// ArgmaxQuantize_4174708212514
// MI455X (gfx1250) — compile-verified, matrix operand order checked by reading, not run on hardware
//
#include <hip/hip_runtime.h>
#include <hip/hip_bf16.h>
#include <stdint.h>

// ---------------------------------------------------------------------------
// Types for CDNA5 WMMA / async copy
// ---------------------------------------------------------------------------
typedef __attribute__((ext_vector_type(16))) __bf16 v16bf;
typedef __attribute__((ext_vector_type(4)))  __bf16 v4bf;
typedef __attribute__((ext_vector_type(8)))  float  v8f;
typedef int v4i __attribute__((vector_size(4 * sizeof(int))));

#define LN_EPS 1e-5f
#define D 512            // feature dim (K)
#define NCODES 8192      // codebook size (N)
#define MROWS 8192       // B*T
#define NSPLIT 4         // N-dimension split across blocks
#define NTILE 32         // columns per staged tile (2 x 16-wide WMMA subtiles)
#define TILE_STRIDE 1056 // bytes per embd row staged in LDS (1024 + 32 pad)

#if defined(__HIP_DEVICE_COMPILE__) && \
    __has_builtin(__builtin_amdgcn_global_load_async_to_lds_b128) && \
    __has_builtin(__builtin_amdgcn_s_wait_asynccnt)
#define ASYNC_OK 1
#else
#define ASYNC_OK 0
#endif

// ---------------------------------------------------------------------------
// Kernel 1: LayerNorm (fp32) -> bf16 rows. One row per wave32.
// ---------------------------------------------------------------------------
__global__ __launch_bounds__(256) void ln_bf16_kernel(
    const float* __restrict__ x, const float* __restrict__ w,
    const float* __restrict__ bia, __bf16* __restrict__ h) {
  const int lane = threadIdx.x & 31;
  const int wave = threadIdx.x >> 5;
  const int row  = blockIdx.x * 8 + wave;           // 1024 blocks * 8 waves = 8192 rows
  const float* xr = x + (size_t)row * D;

  float4 v[4];
  float s = 0.f;
#pragma unroll
  for (int j = 0; j < 4; ++j) {
    v[j] = *(const float4*)(xr + lane * 4 + j * 128);
    s += v[j].x + v[j].y + v[j].z + v[j].w;
  }
#pragma unroll
  for (int off = 16; off >= 1; off >>= 1) s += __shfl_xor(s, off, 32);
  const float mu = s * (1.0f / (float)D);

  float vs = 0.f;
#pragma unroll
  for (int j = 0; j < 4; ++j) {
    float dx = v[j].x - mu, dy = v[j].y - mu, dz = v[j].z - mu, dw = v[j].w - mu;
    vs += dx * dx + dy * dy + dz * dz + dw * dw;
  }
#pragma unroll
  for (int off = 16; off >= 1; off >>= 1) vs += __shfl_xor(vs, off, 32);
  const float rstd = rsqrtf(vs * (1.0f / (float)D) + LN_EPS);

#pragma unroll
  for (int j = 0; j < 4; ++j) {
    const int colb = lane * 4 + j * 128;
    const float4 wv = *(const float4*)(w + colb);
    const float4 bv = *(const float4*)(bia + colb);
    v4bf o;
    o[0] = (__bf16)((v[j].x - mu) * rstd * wv.x + bv.x);
    o[1] = (__bf16)((v[j].y - mu) * rstd * wv.y + bv.y);
    o[2] = (__bf16)((v[j].z - mu) * rstd * wv.z + bv.z);
    o[3] = (__bf16)((v[j].w - mu) * rstd * wv.w + bv.w);
    *(v4bf*)(h + (size_t)row * D + colb) = o;
  }
}

// ---------------------------------------------------------------------------
// Kernel 2: embd fp32 -> bf16 (memory-bound, float4 per thread)
// ---------------------------------------------------------------------------
__global__ __launch_bounds__(256) void cvt_bf16_kernel(
    const float* __restrict__ e, __bf16* __restrict__ o) {
  const size_t g = (size_t)blockIdx.x * blockDim.x + threadIdx.x;  // 1,048,576 threads
  const float4 v = *(const float4*)(e + g * 4);
  v4bf r;
  r[0] = (__bf16)v.x; r[1] = (__bf16)v.y; r[2] = (__bf16)v.z; r[3] = (__bf16)v.w;
  *(v4bf*)(o + g * 4) = r;
}

// ---------------------------------------------------------------------------
// Async tile copy: NTILE embd rows (full K = 1024 B each) -> LDS, padded
// stride. 2048 b128 chunks / 128 threads = 16 per thread.
// ---------------------------------------------------------------------------
__device__ __forceinline__ void copy_tile(const __bf16* __restrict__ e, int n0,
                                          char* __restrict__ buf, int t) {
#pragma unroll
  for (int i = 0; i < (NTILE * 64) / 128; ++i) {
    const int c   = i * 128 + t;
    const int col = c >> 6;       // 0..NTILE-1
    const int wi  = c & 63;       // 0..63 (16-byte chunks within a 1 KB row)
    const char* g = (const char*)e + (((size_t)(n0 + col)) << 10) + (wi << 4);
    char* l = buf + col * TILE_STRIDE + (wi << 4);
#if ASYNC_OK
    __builtin_amdgcn_global_load_async_to_lds_b128(
        (__attribute__((address_space(1))) v4i*)g,
        (__attribute__((address_space(3))) v4i*)l, 0, 0);
#else
    *(float4*)l = *(const float4*)g;
#endif
  }
}

__device__ __forceinline__ void wait_async_done() {
#if ASYNC_OK
  __builtin_amdgcn_s_wait_asynccnt(0);
#endif
}

// ---------------------------------------------------------------------------
// Kernel 3: fused bf16 GEMM (h @ embd^T) + row-wise argmax.
// Block = 128 threads (4 waves). Each wave: M=16 rows, A resident in VGPRs
// (16 chunks of v16bf = 128 VGPRs). Each 32-col tile runs TWO independent
// WMMA accumulation chains (c0/c1) with software-pipelined B loads so the
// XDL pipe never waits on dscnt==0. Double-buffered LDS staging via async
// global->LDS copies (ASYNCcnt).
// ---------------------------------------------------------------------------
__global__ __launch_bounds__(128) void gemm_argmax_kernel(
    const __bf16* __restrict__ h, const __bf16* __restrict__ e,
    float* __restrict__ pmax, int* __restrict__ pidx) {
  __shared__ __align__(64) char smem[2 * NTILE * TILE_STRIDE];  // 66 KB

  const int t     = threadIdx.x;
  const int lane  = t & 31;
  const int wave  = t >> 5;
  const int blk   = blockIdx.x & 127;   // 128 row-blocks of 64 rows
  const int split = blockIdx.x >> 7;    // 4 N-splits
  const int rbase = blk * 64 + wave * 16;
  const int col   = lane & 15;          // N (and M for A load) index within subtile
  const int khalf = (lane >> 4) << 4;   // 0 or 16: contiguous K-half (k = 16h + s), same for A and B

  // --- Load A: 16 rows x K=512 bf16, resident. Lane l%16 = row, l/16 = K half.
  v16bf a[16];
  {
    const __bf16* ap = h + (size_t)(rbase + col) * D + khalf;
#pragma unroll
    for (int kc = 0; kc < 16; ++kc) a[kc] = *(const v16bf*)(ap + kc * 32);
  }

  float best[8];
  int   besti[8];
#pragma unroll
  for (int r = 0; r < 8; ++r) { best[r] = -3.4e38f; besti[r] = 0; }

  const int n0base   = split * (NCODES / NSPLIT);        // 2048 codes per split
  const int numtiles = (NCODES / NSPLIT) / NTILE;        // 64 tiles of 32 cols
  copy_tile(e, n0base, smem, t);                         // prefetch tile 0

#pragma unroll 1
  for (int tile = 0; tile < numtiles; ++tile) {
    char* cur = smem + (tile & 1) * (NTILE * TILE_STRIDE);
    wait_async_done();       // our slice of the current tile is in LDS
    __syncthreads();         // everyone's slice is in LDS; prev readers done
    if (tile + 1 < numtiles)
      copy_tile(e, n0base + (tile + 1) * NTILE,
                smem + ((tile + 1) & 1) * (NTILE * TILE_STRIDE), t);

    // --- two 16x16 logits subtiles, independent accumulators, pipelined B
    const char* bp0 = cur + col * TILE_STRIDE + (khalf << 1);
    const char* bp1 = bp0 + 16 * TILE_STRIDE;
    v16bf b0 = *(const v16bf*)(bp0);
    v16bf b1 = *(const v16bf*)(bp1);
    v8f c0 = {};
    v8f c1 = {};
#pragma unroll
    for (int kc = 0; kc < 16; ++kc) {
      const int kn = (kc + 1) & 15;              // next chunk (wraps harmlessly)
      v16bf nb0 = *(const v16bf*)(bp0 + kn * 64);
      v16bf nb1 = *(const v16bf*)(bp1 + kn * 64);
      c0 = __builtin_amdgcn_wmma_f32_16x16x32_bf16(
          false, a[kc], false, b0, (short)0, c0, false, false);
      c1 = __builtin_amdgcn_wmma_f32_16x16x32_bf16(
          false, a[kc], false, b1, (short)0, c1, false, false);
      b0 = nb0;
      b1 = nb1;
    }

    // --- fused argmax epilogue (strict > keeps first/lowest index;
    //     sub0 cols precede sub1 cols, so update order preserves ties)
    const int n = n0base + tile * NTILE + col;
#pragma unroll
    for (int r = 0; r < 8; ++r)
      if (c0[r] > best[r]) { best[r] = c0[r]; besti[r] = n; }
#pragma unroll
    for (int r = 0; r < 8; ++r)
      if (c1[r] > best[r]) { best[r] = c1[r]; besti[r] = n + 16; }
  }

  // --- cross-lane reduce: lanes 0-15 hold rows rbase+0..7, lanes 16-31 rows +8..15
#pragma unroll
  for (int r = 0; r < 8; ++r) {
    float m = best[r];
    int   mi = besti[r];
#pragma unroll
    for (int off = 8; off >= 1; off >>= 1) {
      const float om = __shfl_xor(m, off, 32);
      const int   oi = __shfl_xor(mi, off, 32);
      if (om > m || (om == m && oi < mi)) { m = om; mi = oi; }
    }
    if ((lane & 15) == 0) {
      const int row = rbase + (lane >> 4) * 8 + r;
      pmax[row * NSPLIT + split] = m;
      pidx[row * NSPLIT + split] = mi;
    }
  }
}

// ---------------------------------------------------------------------------
// Kernel 4: combine partial argmax over the 4 N-splits -> final idx (int32)
// ---------------------------------------------------------------------------
__global__ __launch_bounds__(256) void combine_kernel(
    const float* __restrict__ pmax, const int* __restrict__ pidx,
    int* __restrict__ idx_out) {
  const int m = blockIdx.x * blockDim.x + threadIdx.x;
  if (m >= MROWS) return;
  float best = pmax[m * NSPLIT];
  int   bi   = pidx[m * NSPLIT];
#pragma unroll
  for (int s = 1; s < NSPLIT; ++s) {
    const float v = pmax[m * NSPLIT + s];
    const int   i = pidx[m * NSPLIT + s];
    if (v > best || (v == best && i < bi)) { best = v; bi = i; }
  }
  idx_out[m] = bi;
}

// ---------------------------------------------------------------------------
// Kernel 5: quantize = embd[idx] gather (fp32, float4 coalesced)
// ---------------------------------------------------------------------------
__global__ __launch_bounds__(256) void gather_kernel(
    const float* __restrict__ embd, const int* __restrict__ idx,
    float* __restrict__ out) {
  const int g   = blockIdx.x * blockDim.x + threadIdx.x;  // 1,048,576 threads
  const int row = g >> 7;
  const int off = (g & 127) << 2;
  const float4 v = *(const float4*)(embd + (size_t)idx[row] * D + off);
  *(float4*)(out + (size_t)row * D + off) = v;
}

// ---------------------------------------------------------------------------
// Host launcher
// ---------------------------------------------------------------------------
extern "C" void kernel_launch(void* const* d_in, const int* in_sizes, int n_in,
                              void* d_out, int out_size, void* d_ws, size_t ws_size,
                              hipStream_t stream) {
  const float* x    = (const float*)d_in[0];   // [8,1024,512] fp32
  const float* lnw  = (const float*)d_in[1];   // [512]
  const float* lnb  = (const float*)d_in[2];   // [512]
  const float* embd = (const float*)d_in[3];   // [8192,512] fp32

  float* out = (float*)d_out;                              // quantize: 4,194,304 fp32
  int*   idx_out = (int*)d_out + (size_t)MROWS * D;        // idxs: 8192 int32 tail

  char* ws = (char*)d_ws;
  __bf16* h_bf = (__bf16*)ws;                                   // 8 MB
  __bf16* e_bf = (__bf16*)(ws + (size_t)8 * 1024 * 1024);       // 8 MB
  float*  pmax = (float*)(ws + (size_t)16 * 1024 * 1024);       // 128 KB
  int*    pidx = (int*)(ws + (size_t)16 * 1024 * 1024 + 131072);// 128 KB

  ln_bf16_kernel<<<MROWS / 8, 256, 0, stream>>>(x, lnw, lnb, h_bf);
  cvt_bf16_kernel<<<(NCODES * D / 4) / 256, 256, 0, stream>>>(embd, e_bf);
  gemm_argmax_kernel<<<(MROWS / 64) * NSPLIT, 128, 0, stream>>>(h_bf, e_bf, pmax, pidx);
  combine_kernel<<<MROWS / 256, 256, 0, stream>>>(pmax, pidx, idx_out);
  gather_kernel<<<(MROWS * D / 4) / 256, 256, 0, stream>>>(embd, idx_out, out);
}
